// SSIM_46935402611260
// MI455X (gfx1250) — matrix operand order repaired and checked by reading, not compiled, not run on hardware
//
#include <hip/hip_runtime.h>

typedef _Float16 f16;
typedef __attribute__((ext_vector_type(16))) _Float16 v16h;
typedef __attribute__((ext_vector_type(8)))  float    v8f;

// Normalized 1-D Gaussian, sigma=1.5, 11 taps (outer product == reference 2-D window)
__device__ constexpr float W11c[11] = {
    0.00102840f, 0.00759886f, 0.03600077f, 0.10936070f, 0.21300553f,
    0.26601172f, 0.21300553f, 0.10936070f, 0.03600077f, 0.00759886f, 0.00102840f};

__device__ inline f16 wsel(int d) {
    float r = 0.f;
    #pragma unroll
    for (int j = 0; j < 11; ++j) r = (d == j) ? W11c[j] : r;   // v_cndmask chain, stays in regs
    return (f16)r;
}

// 128-thread block = 4 waves; block owns a 64x64 output tile, each wave a 16x64 strip.
__global__ __launch_bounds__(128) void ssim_wmma_kernel(const float* __restrict__ pred,
                                                        const float* __restrict__ targ,
                                                        float* __restrict__ out) {
    // Shared halo region 74x74 padded to 80x80, stored transposed [col][row].
    // Inner stride 82 halves = 164B = 41 dwords: dword-aligned rows, good bank spread.
    __shared__ f16 sP[80][82];
    __shared__ f16 sT[80][82];
    __shared__ f16 sV[4][16][34];   // per-wave vertical-pass tile, stride 34 halves = 17 dwords

    const int tid  = threadIdx.x;
    const int lane = tid & 31;
    const int w    = tid >> 5;

    const int blk = blockIdx.x;       // 32 images * 8*8 (64x64) tiles = 2048 blocks
    const int b   = blk >> 6;
    const int by  = (blk >> 3) & 7;   // 64-row tile index
    const int bx  = blk & 7;          // 64-col tile index

    const int half = lane >> 4;
    const int m16  = lane & 15;

    // ---------- Stage A: cooperative load of 74x74 halo, luma -> transposed f16 LDS ----------
    {
        const size_t ibase = (size_t)b * (3u * 512u * 512u);
        const float* P = pred + ibase;
        const float* T = targ + ibase;
        for (int idx = tid; idx < 6400; idx += 128) {   // 80*80 padded region
            const int r = idx / 80;
            const int c = idx - r * 80;
            const int gr = by * 64 + r - 5;
            const int gc = bx * 64 + c - 5;
            float p = 0.f, t = 0.f;
            if (r < 74 && c < 74 && (unsigned)gr < 512u && (unsigned)gc < 512u) {
                const size_t o = (size_t)gr * 512 + (size_t)gc;
                p = 0.299f * P[o] + 0.587f * P[o + 262144] + 0.114f * P[o + 524288];
                t = 0.299f * T[o] + 0.587f * T[o + 262144] + 0.114f * T[o + 524288];
            }
            sP[c][r] = (f16)p;   // transposed store
            sT[c][r] = (f16)t;
        }
    }
    __syncthreads();   // sP/sT read-only from here on; sV is wave-private

    // ---------- Constant banded weight fragments ----------
    // aband: A-layout 16x32, A[m][k] = w[k-m]  (vertical pass weights)
    // bband: B-layout 32x16, B[k][x] = w[k-x]  (horizontal pass weights)
    v16h aband, bband;
    #pragma unroll
    for (int v = 0; v < 8; ++v) {
        const int ka = half * 16 + v * 2;
        aband[2 * v]     = wsel(ka - m16);
        aband[2 * v + 1] = wsel(ka + 1 - m16);
        const int kb = (v & 3) * 2 + (v >> 2) * 16 + half * 8;
        bband[2 * v]     = wsel(kb - m16);
        bband[2 * v + 1] = wsel(kb + 1 - m16);
    }

    // Wave w handles output rows [by*64 + w*16, +16); vertical K window starts at region row w*16.
    const int krow0_dw = w * 8;   // dword offset of region-row base for this wave's strip

    const float C1 = 6.5025f, C2 = 58.5225f;
    float sum = 0.f;

    for (int j = 0; j < 4; ++j) {          // four 16-col output tiles in the strip
        v8f D[5];
        #pragma unroll
        for (int i = 0; i < 5; ++i) {      // stats: p, t, p*p, t*t, p*t
            // Pass 1 (vertical): column groups g covering region cols [j*16, j*16+32)
            #pragma unroll
            for (int g = 0; g < 2; ++g) {
                const int cc = j * 16 + g * 16 + m16;           // region column
                const unsigned int* rp = (const unsigned int*)(&sP[cc][0]);
                const unsigned int* rt = (const unsigned int*)(&sT[cc][0]);
                v16h bfrag;
                #pragma unroll
                for (int v = 0; v < 8; ++v) {
                    const int kd = krow0_dw + half * 8 + v;     // K = w*16 + half*16 + v*2 (+0/1)
                    union { unsigned int u; f16 h[2]; } cp, ct;
                    cp.u = rp[kd];
                    ct.u = rt[kd];
                    f16 s0, s1;
                    if      (i == 0) { s0 = cp.h[0];            s1 = cp.h[1]; }
                    else if (i == 1) { s0 = ct.h[0];            s1 = ct.h[1]; }
                    else if (i == 2) { s0 = cp.h[0] * cp.h[0];  s1 = cp.h[1] * cp.h[1]; }
                    else if (i == 3) { s0 = ct.h[0] * ct.h[0];  s1 = ct.h[1] * ct.h[1]; }
                    else             { s0 = cp.h[0] * ct.h[0];  s1 = cp.h[1] * ct.h[1]; }
                    bfrag[2 * v]     = s0;
                    bfrag[2 * v + 1] = s1;
                }
                v8f acc = {};
                acc = __builtin_amdgcn_wmma_f32_16x16x32_f16(false, aband, false, bfrag,
                                                             (short)0, acc, false, false);
                #pragma unroll
                for (int e = 0; e < 8; ++e)                     // D: M = e + half*8, N = m16
                    sV[w][e + half * 8][g * 16 + m16] = (f16)acc[e];
            }

            // Pass 2 (horizontal): out = V(16x32) x Wband(32x16)
            const unsigned int* rv = (const unsigned int*)(&sV[w][m16][0]);
            v16h afrag;
            #pragma unroll
            for (int v = 0; v < 8; ++v) {
                const int kd = (v & 3) + (v >> 2) * 8 + half * 4;
                union { unsigned int u; f16 h[2]; } cv;
                cv.u = rv[kd];
                afrag[2 * v]     = cv.h[0];
                afrag[2 * v + 1] = cv.h[1];
            }
            v8f z = {};
            D[i] = __builtin_amdgcn_wmma_f32_16x16x32_f16(false, afrag, false, bband,
                                                          (short)0, z, false, false);
        }

        // ---------- SSIM for this 16x16 tile ----------
        #pragma unroll
        for (int e = 0; e < 8; ++e) {
            const float mx  = D[0][e], my  = D[1][e];
            const float exx = D[2][e], eyy = D[3][e], exy = D[4][e];
            const float mxy = mx * my, mx2 = mx * mx, my2 = my * my;
            const float n = (2.f * mxy + C1) * (2.f * (exy - mxy) + C2);
            const float d = (mx2 + my2 + C1) * ((exx - mx2) + (eyy - my2) + C2);
            sum += n / d;
        }
    }

    // ---------- Wave reduction + single atomic per wave ----------
    #pragma unroll
    for (int off = 16; off > 0; off >>= 1)
        sum += __shfl_xor(sum, off, 32);
    if (lane == 0)
        atomicAdd(out, sum * (1.0f / 8388608.0f));   // 1 / (32*512*512)
}

__global__ void ssim_zero_out(float* out) {
    if (threadIdx.x == 0) out[0] = 0.f;
}

extern "C" void kernel_launch(void* const* d_in, const int* in_sizes, int n_in,
                              void* d_out, int out_size, void* d_ws, size_t ws_size,
                              hipStream_t stream) {
    const float* pred = (const float*)d_in[0];
    const float* targ = (const float*)d_in[1];
    float* out = (float*)d_out;
    (void)in_sizes; (void)n_in; (void)out_size; (void)d_ws; (void)ws_size;

    ssim_zero_out<<<1, 32, 0, stream>>>(out);
    // 32 images * (512/64)^2 tiles = 2048 blocks of 128 threads (4 waves)
    ssim_wmma_kernel<<<2048, 128, 0, stream>>>(pred, targ, out);
}
